// Interacte_GRU_29119878267029
// MI455X (gfx1250) — compile-verified, matrix operand order checked by reading, not run on hardware
//
#include <hip/hip_runtime.h>
#include <cstdint>
#include <cstddef>

// ---------------------------------------------------------------------------
// Problem dimensions
// ---------------------------------------------------------------------------
#define N_B   256
#define N_T   256
#define N_IN  256
#define N_HID 512
#define N_OUT 256
#define ROWS  (N_B * N_T)          // 65536 rows for the pre-projection GEMM
#define KCAT  (N_IN * 2)           // 512 : concat(x, c) along K

#define GRU_BLOCKS 32
#define GRU_THREADS 256

// ---------------------------------------------------------------------------
// WMMA types (CDNA5 gfx1250, wave32): v_wmma_f32_16x16x32_bf16
// ---------------------------------------------------------------------------
typedef __attribute__((ext_vector_type(16))) __bf16 v16bf;
typedef __attribute__((ext_vector_type(8)))  float  v8f;
typedef unsigned int v4u __attribute__((ext_vector_type(4)));

union Frag {
  v16bf v;
  v4u   q[2];
};

__device__ __forceinline__ v8f wmma_bf16(const Frag& a, const Frag& b, v8f c) {
  return __builtin_amdgcn_wmma_f32_16x16x32_bf16(
      /*neg_a=*/false, a.v, /*neg_b=*/false, b.v,
      /*c_mod=*/(short)0, c, /*reuse_a=*/false, /*reuse_b=*/false);
}

__device__ __forceinline__ unsigned short f2bf(float f) {
  unsigned u = __float_as_uint(f);
  u += 0x7FFFu + ((u >> 16) & 1u);   // round-to-nearest-even
  return (unsigned short)(u >> 16);
}

__device__ __forceinline__ float sigmoidf_(float x) {
  return 1.0f / (1.0f + __expf(-x));
}

// A fragment (16x32 bf16, row-major source, leading dim ldk).
// Lanes 0-15 hold K = [0..7, 16..23]; lanes 16-31 hold K = [8..15, 24..31].
__device__ __forceinline__ Frag load_a_frag(const unsigned short* A, int row,
                                            int k0, int lane, int ldk) {
  const unsigned short* p = A + (size_t)row * ldk + k0 + ((lane >> 4) << 3);
  Frag f;
  f.q[0] = *(const v4u*)(p);
  f.q[1] = *(const v4u*)(p + 16);
  return f;
}

// B fragment from fragment-major packed weights:
// packed index = (((kt*NT + nt)*32 + lane)*16 + h) -> 32B contiguous per lane.
__device__ __forceinline__ Frag load_b_frag(const unsigned short* W, int kt,
                                            int nt, int NT, int lane) {
  const unsigned short* p = W + ((((size_t)kt * NT + nt) * 32 + lane) << 4);
  Frag f;
  f.q[0] = *(const v4u*)(p);
  f.q[1] = *(const v4u*)(p + 8);
  return f;
}

// ---------------------------------------------------------------------------
// Workspace layout (bytes). Total ~474 MB.
// ---------------------------------------------------------------------------
static constexpr size_t SZ_XC   = (size_t)ROWS * KCAT * 2;
static constexpr size_t SZ_WPRE = (size_t)3 * KCAT * N_HID * 2;
static constexpr size_t SZ_WHID = (size_t)3 * N_HID * N_HID * 2;
static constexpr size_t SZ_WOUT = (size_t)N_HID * N_OUT * 2;
static constexpr size_t SZ_PRE  = (size_t)3 * N_T * N_B * N_HID * 4;
static constexpr size_t SZ_H32  = (size_t)N_B * N_HID * 4;
static constexpr size_t SZ_H16  = (size_t)N_B * N_HID * 2;
static constexpr size_t SZ_RH   = SZ_H16;
static constexpr size_t SZ_Z    = SZ_H32;

static constexpr size_t OFF_XC   = 0;
static constexpr size_t OFF_WPRE = OFF_XC   + SZ_XC;
static constexpr size_t OFF_WHID = OFF_WPRE + SZ_WPRE;
static constexpr size_t OFF_WOUT = OFF_WHID + SZ_WHID;
static constexpr size_t OFF_PRE  = OFF_WOUT + SZ_WOUT;
static constexpr size_t OFF_H32  = OFF_PRE  + SZ_PRE;
static constexpr size_t OFF_H16  = OFF_H32  + SZ_H32;
static constexpr size_t OFF_RH   = OFF_H16  + SZ_H16;
static constexpr size_t OFF_Z    = OFF_RH   + SZ_RH;
static constexpr size_t OFF_BAR  = OFF_Z    + SZ_Z;

// ---------------------------------------------------------------------------
// Prep kernels
// ---------------------------------------------------------------------------
__global__ __launch_bounds__(256) void convert_xc_kernel(const float* __restrict__ x,
                                                         const float* __restrict__ c,
                                                         unsigned short* __restrict__ XC) {
  size_t stride = (size_t)gridDim.x * blockDim.x;
  for (size_t i = (size_t)blockIdx.x * blockDim.x + threadIdx.x;
       i < (size_t)ROWS * KCAT; i += stride) {
    int k = (int)(i & (KCAT - 1));
    size_t row = i >> 9;
    float v = (k < N_IN) ? x[row * N_IN + k] : c[row * N_IN + (k - N_IN)];
    XC[i] = f2bf(v);
  }
}

__global__ __launch_bounds__(256) void pack_w_kernel(const float* __restrict__ src0,
                                                     const float* __restrict__ src1,
                                                     int K0, int K, int N,
                                                     unsigned short* __restrict__ dst) {
  int NT = N >> 4;
  size_t total = (size_t)(K >> 5) * NT * 512;
  size_t stride = (size_t)gridDim.x * blockDim.x;
  for (size_t i = (size_t)blockIdx.x * blockDim.x + threadIdx.x; i < total; i += stride) {
    int h    = (int)(i & 15);
    int lane = (int)((i >> 4) & 31);
    size_t tile = i >> 9;
    int nt = (int)(tile % NT);
    int kt = (int)(tile / NT);
    int n  = nt * 16 + (lane & 15);
    int kb = (lane < 16) ? 0 : 8;
    int kin = (h < 8) ? (kb + h) : (16 + kb + (h - 8));
    int k = kt * 32 + kin;
    float v = (k < K0) ? src0[(size_t)k * N + n] : src1[(size_t)(k - K0) * N + n];
    dst[i] = f2bf(v);
  }
}

__global__ __launch_bounds__(256) void init_h_kernel(const float* __restrict__ h0,
                                                     float* __restrict__ h32,
                                                     unsigned short* __restrict__ h16,
                                                     unsigned* __restrict__ bar) {
  size_t i = (size_t)blockIdx.x * blockDim.x + threadIdx.x;
  if (i < (size_t)N_B * N_HID) {
    float v = h0[i];
    h32[i] = v;
    h16[i] = f2bf(v);
  }
  if (blockIdx.x == 0 && threadIdx.x < 16) bar[threadIdx.x] = 0;
}

// ---------------------------------------------------------------------------
// Phase 1: pre_g[t][b][:] = [x|c] @ Wg + b_g  for g in {z, r, h}
// Block = 128(M) x 64(N) tile, 8 waves; A+B staged via async global->LDS.
// ---------------------------------------------------------------------------
__global__ __launch_bounds__(256) void gemm_pre_kernel(
    const unsigned short* __restrict__ XC,
    const unsigned short* __restrict__ Wpre,
    const float* __restrict__ b_r, const float* __restrict__ b_z,
    const float* __restrict__ b_h,
    float* __restrict__ pre)
{
  __shared__ __align__(16) unsigned short sm[(8192 + 4096) / 2];  // A:8KB B:4KB

  const int tid  = threadIdx.x;
  const int wave = tid >> 5;
  const int lane = tid & 31;

  // Escape the LDS array's address into the asm operands (ptrtoint capture),
  // so the "memory"-clobbering async-copy asm is a potential writer of sm and
  // the ds_load reads below cannot be folded to undef.
  unsigned smbase = (unsigned)(uintptr_t)(void*)sm;   // flat low 32 bits == LDS offset
  if (smbase == 0xFFFFFFFFu) sm[0] = 0;               // opaque, never taken

  int bid = blockIdx.x;
  int g   = bid / 4096;           // 0=z, 1=r, 2=h
  int rem = bid % 4096;
  int mt  = rem >> 3;
  int ns  = rem & 7;
  int m0  = mt * 128;
  int n0  = ns * 64;
  int nt0 = n0 >> 4;

  const unsigned short* Wg = Wpre + (size_t)g * (KCAT * N_HID);
  const float* bias = (g == 0) ? b_z : (g == 1) ? b_r : b_h;
  float* preg = pre + (size_t)g * ((size_t)N_T * N_B * N_HID);

  v8f acc[4] = {v8f{}, v8f{}, v8f{}, v8f{}};

  // Precomputed per-thread chunk geometry for staging.
  int rowA0 = tid >> 2, offA0 = tid & 3;            // chunk tid
  int rowA1 = (tid + 256) >> 2, offA1 = tid & 3;    // chunk tid+256

  for (int kt = 0; kt < 16; ++kt) {
    // A tile: 128 rows x 32 bf16 = 8 KB, 512 chunks of 16 B (2 per thread)
    {
      const void* gp0 =
          (const void*)(XC + ((size_t)(m0 + rowA0) * KCAT + kt * 32 + offA0 * 8));
      unsigned l0 = smbase + (unsigned)(tid * 16);
      asm volatile("global_load_async_to_lds_b128 %0, %1, off"
                   :: "v"(l0), "v"(gp0) : "memory");
      const void* gp1 =
          (const void*)(XC + ((size_t)(m0 + rowA1) * KCAT + kt * 32 + offA1 * 8));
      unsigned l1 = smbase + (unsigned)((tid + 256) * 16);
      asm volatile("global_load_async_to_lds_b128 %0, %1, off"
                   :: "v"(l1), "v"(gp1) : "memory");
    }
    // B tile: 4 n-tiles x 1 KB contiguous in packed W (1 chunk per thread)
    {
      const void* gp =
          (const void*)(Wg + (((size_t)(kt * 32 + nt0)) << 9) + tid * 8);
      unsigned l2 = smbase + 8192u + (unsigned)(tid * 16);
      asm volatile("global_load_async_to_lds_b128 %0, %1, off"
                   :: "v"(l2), "v"(gp) : "memory");
    }
    asm volatile("s_wait_asynccnt 0" ::: "memory");
    __syncthreads();

    // A fragment from LDS (elements): row*32 + klo
    int row = wave * 16 + (lane & 15);
    const unsigned short* pa = sm + row * 32 + ((lane >> 4) << 3);
    Frag a;
    a.q[0] = *(const v4u*)(pa);
    a.q[1] = *(const v4u*)(pa + 16);

    const unsigned short* sB = sm + 4096;  // B region (elements)
#pragma unroll
    for (int j = 0; j < 4; ++j) {
      const unsigned short* pb = sB + j * 512 + lane * 16;
      Frag b;
      b.q[0] = *(const v4u*)(pb);
      b.q[1] = *(const v4u*)(pb + 8);
      acc[j] = wmma_bf16(a, b, acc[j]);
    }
    __syncthreads();
  }

  // Epilogue: add bias, scatter to [T][B][HID]
#pragma unroll
  for (int j = 0; j < 4; ++j) {
#pragma unroll
    for (int v = 0; v < 8; ++v) {
      int Rg = m0 + wave * 16 + v + ((lane >> 4) << 3);  // flattened b*T + t
      int col = n0 + j * 16 + (lane & 15);
      int b = Rg >> 8;
      int t = Rg & 255;
      preg[((size_t)t * N_B + b) * N_HID + col] = acc[j][v] + bias[col];
    }
  }
}

// ---------------------------------------------------------------------------
// Persistent GRU scan with global grid barrier
// ---------------------------------------------------------------------------
__device__ __forceinline__ void grid_sync(unsigned* bar) {
  __syncthreads();
  if (threadIdx.x == 0) {
    __threadfence();
    unsigned gen = atomicAdd(&bar[1], 0u);
    if (atomicAdd(&bar[0], 1u) == (unsigned)(GRU_BLOCKS - 1)) {
      atomicExch(&bar[0], 0u);
      __threadfence();
      atomicAdd(&bar[1], 1u);
    } else {
      while (atomicAdd(&bar[1], 0u) == gen) { __builtin_amdgcn_s_sleep(2); }
    }
    __threadfence();
  }
  __syncthreads();
}

__global__ __launch_bounds__(GRU_THREADS) void gru_kernel(
    const unsigned short* __restrict__ Whid,
    const float* __restrict__ pre,
    float* __restrict__ h32, unsigned short* __restrict__ h16,
    unsigned short* __restrict__ rh16, float* __restrict__ zbuf,
    float* __restrict__ hs,
    unsigned* __restrict__ bar)
{
  const int tid  = threadIdx.x;
  const int lane = tid & 31;
  const int gw   = blockIdx.x * (GRU_THREADS / 32) + (tid >> 5);  // 0..255

  const float* pre_z = pre;
  const float* pre_r = pre + (size_t)N_T * N_B * N_HID;
  const float* pre_h = pre + (size_t)2 * N_T * N_B * N_HID;

  for (int t = 0; t < N_T; ++t) {
    // Phase A: z = sigmoid(pre_z + h@Whz); rh = sigmoid(pre_r + h@Whr) * h
    {
      int mat = gw >> 7;                 // 0 -> z, 1 -> r
      int rem = gw & 127;
      int mt = rem >> 3, ns = rem & 7;
      int m0 = mt * 16, n0 = ns * 64, nt0 = n0 >> 4;
      const unsigned short* W = Whid + (size_t)mat * (N_HID * N_HID);
      v8f acc[4] = {v8f{}, v8f{}, v8f{}, v8f{}};
      int row = m0 + (lane & 15);
#pragma unroll 2
      for (int kt = 0; kt < 16; ++kt) {
        Frag a = load_a_frag(h16, row, kt * 32, lane, N_HID);
#pragma unroll
        for (int j = 0; j < 4; ++j) {
          Frag b = load_b_frag(W, kt, nt0 + j, 32, lane);
          acc[j] = wmma_bf16(a, b, acc[j]);
        }
      }
      const float* pg = ((mat == 0) ? pre_z : pre_r) + (size_t)t * N_B * N_HID;
#pragma unroll
      for (int j = 0; j < 4; ++j) {
#pragma unroll
        for (int v = 0; v < 8; ++v) {
          int R = m0 + v + ((lane >> 4) << 3);
          int nn = n0 + j * 16 + (lane & 15);
          size_t idx = (size_t)R * N_HID + nn;
          float s = sigmoidf_(pg[idx] + acc[j][v]);
          if (mat == 0) zbuf[idx] = s;
          else          rh16[idx] = f2bf(s * h32[idx]);
        }
      }
    }
    grid_sync(bar);

    // Phase B: h_new = (1-z)*h + z*tanh(pre_h + (r*h)@Whh)
    if (gw < 128) {
      int mt = gw >> 3, ns = gw & 7;
      int m0 = mt * 16, n0 = ns * 64, nt0 = n0 >> 4;
      const unsigned short* W = Whid + (size_t)2 * (N_HID * N_HID);
      v8f acc[4] = {v8f{}, v8f{}, v8f{}, v8f{}};
      int row = m0 + (lane & 15);
#pragma unroll 2
      for (int kt = 0; kt < 16; ++kt) {
        Frag a = load_a_frag(rh16, row, kt * 32, lane, N_HID);
#pragma unroll
        for (int j = 0; j < 4; ++j) {
          Frag b = load_b_frag(W, kt, nt0 + j, 32, lane);
          acc[j] = wmma_bf16(a, b, acc[j]);
        }
      }
      const float* ph = pre_h + (size_t)t * N_B * N_HID;
#pragma unroll
      for (int j = 0; j < 4; ++j) {
#pragma unroll
        for (int v = 0; v < 8; ++v) {
          int R = m0 + v + ((lane >> 4) << 3);
          int nn = n0 + j * 16 + (lane & 15);
          size_t idx = (size_t)R * N_HID + nn;
          float ht = tanhf(ph[idx] + acc[j][v]);
          float z = zbuf[idx];
          float hn = (1.0f - z) * h32[idx] + z * ht;
          h32[idx] = hn;
          h16[idx] = f2bf(hn);
          hs[((size_t)R * N_T + t) * N_HID + nn] = hn;
        }
      }
    }
    grid_sync(bar);
  }
}

// ---------------------------------------------------------------------------
// Output head: relu(h_last @ w_out + b_out) -> [B][OUT]
// A (32 rows of h_last) staged into LDS via the Tensor Data Mover (TDM).
// ---------------------------------------------------------------------------
__global__ __launch_bounds__(256) void out_kernel(
    const unsigned short* __restrict__ h16,
    const unsigned short* __restrict__ Wout,
    const float* __restrict__ b_out,
    float* __restrict__ out)
{
  __shared__ __align__(16) unsigned short smo[32 * N_HID];  // 32 KB

  const int tid  = threadIdx.x;
  const int lane = tid & 31;
  const int gw   = blockIdx.x * 8 + (tid >> 5);   // 0..63
  int mt = gw >> 2, ns = gw & 3;
  int m0 = mt * 16, n0 = ns * 64, nt0 = n0 >> 4;
  int rowbase = 32 * blockIdx.x;                  // block's 32-row slab

  unsigned smbase = (unsigned)(uintptr_t)(void*)smo;
  if (smbase == 0xFFFFFFFFu) smo[0] = 0;          // opaque, never taken

#if __has_builtin(__builtin_amdgcn_tensor_load_to_lds)
  if (tid < 32) {  // one wave issues the TDM descriptor for the block
    typedef unsigned int u32x4 __attribute__((ext_vector_type(4)));
    typedef int i32x4 __attribute__((ext_vector_type(4)));
    typedef int i32x8 __attribute__((ext_vector_type(8)));
    unsigned long long ga =
        (unsigned long long)(const void*)(h16 + (size_t)rowbase * N_HID);
    u32x4 g0;
    g0.x = 1u;                                        // count=1, user mode
    g0.y = smbase;                                    // lds_addr
    g0.z = (unsigned)(ga & 0xFFFFFFFFu);              // global_addr[31:0]
    g0.w = (unsigned)((ga >> 32) & 0x1FFFFFFu) | (2u << 30);  // addr[56:32]|type=2
    i32x8 g1;
    g1[0] = (int)(1u << 16);        // data_size=1 (2 bytes), mask=0
    g1[1] = (int)(512u << 16);      // tensor_dim0 = 512 (low 16 in [63:48])
    g1[2] = (int)(32u << 16);       // dim0 hi=0 | tensor_dim1 = 32
    g1[3] = (int)(512u << 16);      // dim1 hi=0 | tile_dim0 = 512
    g1[4] = 32;                     // tile_dim1 = 32, tile_dim2 = 0
    g1[5] = 512;                    // tensor_dim0_stride = 512
    g1[6] = 0;
    g1[7] = 0;
    i32x4 gz = {0, 0, 0, 0};
#if defined(__clang_major__) && (__clang_major__ >= 23)
    i32x8 gz8 = {0, 0, 0, 0, 0, 0, 0, 0};
    __builtin_amdgcn_tensor_load_to_lds(g0, g1, gz, gz, gz8, 0);
#else
    __builtin_amdgcn_tensor_load_to_lds(g0, g1, gz, gz, 0);
#endif
#if __has_builtin(__builtin_amdgcn_s_wait_tensorcnt)
    __builtin_amdgcn_s_wait_tensorcnt(0);
#else
    asm volatile("s_wait_tensorcnt 0" ::: "memory");
#endif
  }
  __syncthreads();
#else
  // Fallback: cooperative LDS staging with plain loads/stores.
  for (int i = tid; i < 32 * N_HID / 8; i += 256) {
    ((v4u*)smo)[i] = ((const v4u*)(h16 + (size_t)rowbase * N_HID))[i];
  }
  __syncthreads();
#endif

  v8f acc[4] = {v8f{}, v8f{}, v8f{}, v8f{}};
  int rl = (m0 - rowbase) + (lane & 15);   // row within the staged slab
#pragma unroll 2
  for (int kt = 0; kt < 16; ++kt) {
    const unsigned short* pa = smo + (size_t)rl * N_HID + kt * 32 + ((lane >> 4) << 3);
    Frag a;
    a.q[0] = *(const v4u*)(pa);
    a.q[1] = *(const v4u*)(pa + 16);
#pragma unroll
    for (int j = 0; j < 4; ++j) {
      Frag b = load_b_frag(Wout, kt, nt0 + j, 16, lane);
      acc[j] = wmma_bf16(a, b, acc[j]);
    }
  }
#pragma unroll
  for (int j = 0; j < 4; ++j) {
#pragma unroll
    for (int v = 0; v < 8; ++v) {
      int R = m0 + v + ((lane >> 4) << 3);
      int nn = n0 + j * 16 + (lane & 15);
      float o = acc[j][v] + b_out[nn];
      out[(size_t)R * N_OUT + nn] = (o > 0.0f) ? o : 0.0f;
    }
  }
}

// ---------------------------------------------------------------------------
// Launch
// ---------------------------------------------------------------------------
extern "C" void kernel_launch(void* const* d_in, const int* in_sizes, int n_in,
                              void* d_out, int out_size, void* d_ws, size_t ws_size,
                              hipStream_t stream) {
  (void)in_sizes; (void)n_in; (void)out_size; (void)ws_size;

  const float* x     = (const float*)d_in[0];
  const float* c     = (const float*)d_in[1];
  const float* h0    = (const float*)d_in[2];
  const float* w_xr  = (const float*)d_in[3];
  const float* w_hr  = (const float*)d_in[4];
  const float* w_cr  = (const float*)d_in[5];
  const float* w_xz  = (const float*)d_in[6];
  const float* w_hz  = (const float*)d_in[7];
  const float* w_cz  = (const float*)d_in[8];
  const float* w_xh  = (const float*)d_in[9];
  const float* w_hh  = (const float*)d_in[10];
  const float* w_ch  = (const float*)d_in[11];
  const float* b_r   = (const float*)d_in[12];
  const float* b_z   = (const float*)d_in[13];
  const float* b_h   = (const float*)d_in[14];
  const float* w_out = (const float*)d_in[15];
  const float* b_out = (const float*)d_in[16];

  char* ws = (char*)d_ws;
  unsigned short* XC   = (unsigned short*)(ws + OFF_XC);
  unsigned short* Wpre = (unsigned short*)(ws + OFF_WPRE);
  unsigned short* Whid = (unsigned short*)(ws + OFF_WHID);
  unsigned short* Wout = (unsigned short*)(ws + OFF_WOUT);
  float*          pre  = (float*)(ws + OFF_PRE);
  float*          h32  = (float*)(ws + OFF_H32);
  unsigned short* h16  = (unsigned short*)(ws + OFF_H16);
  unsigned short* rh16 = (unsigned short*)(ws + OFF_RH);
  float*          zbuf = (float*)(ws + OFF_Z);
  unsigned*       bar  = (unsigned*)(ws + OFF_BAR);

  float* out = (float*)d_out;
  float* hs  = out + (size_t)N_B * N_OUT;

  const size_t WMAT = (size_t)KCAT * N_HID;

  convert_xc_kernel<<<32768, 256, 0, stream>>>(x, c, XC);
  pack_w_kernel<<<1024, 256, 0, stream>>>(w_xz, w_cz, N_IN, KCAT, N_HID, Wpre + 0 * WMAT);
  pack_w_kernel<<<1024, 256, 0, stream>>>(w_xr, w_cr, N_IN, KCAT, N_HID, Wpre + 1 * WMAT);
  pack_w_kernel<<<1024, 256, 0, stream>>>(w_xh, w_ch, N_IN, KCAT, N_HID, Wpre + 2 * WMAT);
  pack_w_kernel<<<1024, 256, 0, stream>>>(w_hz, w_hz, N_HID, N_HID, N_HID, Whid + 0 * WMAT);
  pack_w_kernel<<<1024, 256, 0, stream>>>(w_hr, w_hr, N_HID, N_HID, N_HID, Whid + 1 * WMAT);
  pack_w_kernel<<<1024, 256, 0, stream>>>(w_hh, w_hh, N_HID, N_HID, N_HID, Whid + 2 * WMAT);
  pack_w_kernel<<<512, 256, 0, stream>>>(w_out, w_out, N_HID, N_HID, N_OUT, Wout);
  init_h_kernel<<<(N_B * N_HID + 255) / 256, 256, 0, stream>>>(h0, h32, h16, bar);

  gemm_pre_kernel<<<3 * 512 * 8, 256, 0, stream>>>(XC, Wpre, b_r, b_z, b_h, pre);
  gru_kernel<<<GRU_BLOCKS, GRU_THREADS, 0, stream>>>(Whid, pre, h32, h16, rh16,
                                                     zbuf, hs, bar);
  out_kernel<<<8, 256, 0, stream>>>(h16, Wout, b_out, out);
}
